// HybridBayesQuantum_12850542150426
// MI455X (gfx1250) — matrix operand order repaired and checked by reading, not compiled, not run on hardware
//
#include <hip/hip_runtime.h>

// ---------------------------------------------------------------------------
// HybridBayesQuantum: ResNet18 (bf16 WMMA implicit GEMM, async weight staging)
// -> FC head -> 4-qubit state-vector simulator. MI455X / gfx1250, wave32.
// ---------------------------------------------------------------------------

#define PI_F 3.14159265358979323846f

typedef __attribute__((ext_vector_type(16))) __bf16        v16bf;
typedef __attribute__((ext_vector_type(8)))  float         v8f;
typedef __attribute__((ext_vector_type(4)))  unsigned int  v4u;

union FragBF { v4u u[2]; v16bf v; };   // 32 bytes: one 16x32-bf16 WMMA operand per lane

__device__ __forceinline__ unsigned short f2bf(float f) {
  union { float f; unsigned u; } x; x.f = f;
  unsigned r = x.u + 0x7FFFu + ((x.u >> 16) & 1u);   // round-to-nearest-even
  return (unsigned short)(r >> 16);
}
__device__ __forceinline__ float bf2f(unsigned short h) {
  union { unsigned u; float f; } x; x.u = ((unsigned)h) << 16;
  return x.f;
}

// ---------------------------------------------------------------------------
// Generic implicit-GEMM conv (+BN, +optional bias, +optional residual, +ReLU)
// A = im2col(activations) [M x K], B = weights [K x N], all bf16, f32 accum.
// M = 64*OH*OW, N = Cout, K = Cin*KH*KH  (all multiples of the tile sizes).
// Block = 256 threads (8 waves). Block tile 64M x TNv N (TNv = 64 or 128);
// wave tile 16M x (TNv/2)N -> NF = TNv/32 WMMAs per wave per k-step.
// The wide-N tile amortizes the A-tile im2col gather (the true bottleneck)
// over 2x the FLOPs for all Cout>=128 layers (~70% of network FLOPs).
// B tile staged L2 -> LDS with GLOBAL_LOAD_ASYNC_TO_LDS_B128 (ASYNCcnt),
// overlapping the VALU im2col gather for the A tile.
// ---------------------------------------------------------------------------
struct ConvP {
  const unsigned short* in;     // bf16 activations, NCHW (H==W)
  const unsigned short* w;      // bf16 weights [Cout][Cin*KH*KW]
  const float *bng, *bnb, *bnm, *bnv;
  const float* bias;            // optional pre-BN bias (FC1), else nullptr
  const unsigned short* res;    // optional residual (bf16, same layout as out)
  unsigned short* out;          // bf16, NCHW
  int Cin, H, Cout, OH, relu;
};

#define TM 64
#define TK 32
// LDS: As at [0,4096), Bs at [4096, 4096 + TNv*64)

template <int KH, int STRIDE, int PAD, int TNv>
__global__ __launch_bounds__(256)
void conv_wmma_k(ConvP p) {
  extern __shared__ char smem[];
  unsigned short* As = (unsigned short*)smem;            // 64 x 32 bf16
  unsigned short* Bs = (unsigned short*)(smem + 4096);   // TNv x 32 bf16

  constexpr int KK = KH * KH;
  constexpr int NW = TNv / 2;         // per-wave N width
  constexpr int NF = NW / 16;         // B fragments / WMMAs per wave
  const int K   = p.Cin * KK;
  const int H   = p.H;
  const int OH  = p.OH;
  const int OHW = OH * OH;
  const int HW  = H * H;
  const int m0  = blockIdx.x * TM;
  const int n0  = blockIdx.y * TNv;
  const int tid  = threadIdx.x;
  const int lane = tid & 31;
  const int wave = tid >> 5;

  // WMMA lane mapping (ISA 7.12.2): A lane holds row M=lane%16,
  // K runs [ka,ka+8) and [ka+16,ka+24) with ka = (lane>=16)*8.
  // B lane holds col N=lane%16, K runs [kb,kb+8) and [kb+16,kb+24) with kb = (lane>=16)*8.
  const int fr = lane & 15;
  const int ka = (lane >> 4) << 3;
  const int kb = (lane >> 4) << 3;
  const int wm = (wave >> 1) * 16;    // wave M offset within block tile
  const int wn = (wave & 1) * NW;     // wave N offset within block tile

  v8f acc[NF];
#pragma unroll
  for (int f = 0; f < NF; ++f) acc[f] = (v8f){0.f,0.f,0.f,0.f,0.f,0.f,0.f,0.f};

  // Cooperative tile load: thread covers 8 K-elems of one row.
  const int trow = tid >> 2;          // 0..63
  const int tkk  = (tid & 3) * 8;     // 0,8,16,24
  const int am   = m0 + trow;
  const int img  = am / OHW;
  const int sp   = am - img * OHW;
  const int oh   = sp / OH;
  const int ow   = sp - oh * OH;
  const size_t imgbase = (size_t)img * p.Cin * HW;
  const int ihb = oh * STRIDE - PAD;
  const int iwb = ow * STRIDE - PAD;

  // per-thread constants for the async B stage (16B per lane per issue)
  const unsigned b_voff = (unsigned)((trow * K + tkk) * 2);

  for (int ks = 0; ks < K; ks += TK) {
    __syncthreads();
    // ---- B tile: async DMA global -> LDS (no VGPR data path) ---------------
#pragma unroll
    for (int rep = 0; rep < TNv / 64; ++rep) {
      unsigned lds = 4096u + (unsigned)(((rep * 64 + trow) * TK + tkk) * 2);
      unsigned long long wbase = (unsigned long long)(uintptr_t)
          (p.w + (size_t)(n0 + rep * 64) * K + ks);
      asm volatile("global_load_async_to_lds_b128 %0, %1, %2"
                   :: "v"(lds), "v"(b_voff), "s"(wbase) : "memory");
    }
    // ---- A tile: im2col gather, 8 bf16 per thread (overlaps async B) -------
#pragma unroll
    for (int e = 0; e < 8; ++e) {
      int k  = ks + tkk + e;
      int ci = k / KK;                 // compile-time KK -> magic multiply
      int r  = k - ci * KK;
      int kh = r / KH;
      int kw = r - kh * KH;
      int ih = ihb + kh;
      int iw = iwb + kw;
      unsigned short bits;
      if (PAD == 0 ||
          ((unsigned)ih < (unsigned)H && (unsigned)iw < (unsigned)H))
        bits = p.in[imgbase + (size_t)ci * HW + ih * H + iw];
      else
        bits = 0;
      As[trow * TK + tkk + e] = bits;
    }
    asm volatile("s_wait_asynccnt 0x0" ::: "memory");
    __syncthreads();
    // ---- fragments + NF x WMMA --------------------------------------------
    FragBF fa;
    fa.u[0] = *(const v4u*)&As[(wm + fr) * TK + ka];
    fa.u[1] = *(const v4u*)&As[(wm + fr) * TK + ka + 16];
#pragma unroll
    for (int f = 0; f < NF; ++f) {
      FragBF fb;
      fb.u[0] = *(const v4u*)&Bs[(wn + f * 16 + fr) * TK + kb];
      fb.u[1] = *(const v4u*)&Bs[(wn + f * 16 + fr) * TK + kb + 16];
      acc[f] = __builtin_amdgcn_wmma_f32_16x16x32_bf16(false, fa.v, false, fb.v,
                                                       (short)0, acc[f], false, false);
    }
  }

  // ---- epilogue: BN (+bias), +residual, ReLU, bf16 store -------------------
  // D layout: lane holds N = lane%16, rows M = (lane/16)*8 + r (VGPR r).
  const int en  = lane & 15;
  const int em0 = (lane >> 4) * 8;
#pragma unroll
  for (int f = 0; f < NF; ++f) {
    int co = n0 + wn + f * 16 + en;
    float scale = p.bng[co] * rsqrtf(p.bnv[co] + 1e-5f);
    float shift = p.bnb[co] - p.bnm[co] * scale;
    if (p.bias) shift += p.bias[co] * scale;
#pragma unroll
    for (int r = 0; r < 8; ++r) {
      int m  = m0 + wm + em0 + r;
      int mi = m / OHW;
      int ms = m - mi * OHW;
      size_t ofs = ((size_t)mi * p.Cout + co) * OHW + ms;
      float v = acc[f][r] * scale + shift;
      if (p.res)  v += bf2f(p.res[ofs]);
      if (p.relu) v  = fmaxf(v, 0.f);
      p.out[ofs] = f2bf(v);
    }
  }
}

// ---------------------------------------------------------------------------
// conv1: 7x7 s2 p3, Cin=3 (K=147, not a multiple of 32) -> direct fp32 + BN/ReLU
// ---------------------------------------------------------------------------
__global__ void conv1_k(const float* __restrict__ x, const float* __restrict__ w,
                        const float* g, const float* b, const float* m, const float* v,
                        unsigned short* out) {
  size_t idx = (size_t)blockIdx.x * 256 + threadIdx.x;   // [64,64,112,112]
  int ow = (int)(idx % 112); size_t t = idx / 112;
  int oh = (int)(t % 112);   t /= 112;
  int co = (int)(t % 64);
  int img = (int)(t / 64);
  float acc = 0.f;
  for (int ci = 0; ci < 3; ++ci)
    for (int kh = 0; kh < 7; ++kh) {
      int ih = oh * 2 + kh - 3;
      if ((unsigned)ih >= 224u) continue;
      for (int kw = 0; kw < 7; ++kw) {
        int iw = ow * 2 + kw - 3;
        if ((unsigned)iw >= 224u) continue;
        acc += x[(((size_t)img * 3 + ci) * 224 + ih) * 224 + iw] *
               w[((co * 3 + ci) * 7 + kh) * 7 + kw];
      }
    }
  float scale = g[co] * rsqrtf(v[co] + 1e-5f);
  float val = (acc - m[co]) * scale + b[co];
  out[idx] = f2bf(fmaxf(val, 0.f));
}

// 3x3 s2 p1 maxpool: [64,64,112,112] -> [64,64,56,56], bf16
__global__ void maxpool_k(const unsigned short* __restrict__ in, unsigned short* out) {
  size_t idx = (size_t)blockIdx.x * 256 + threadIdx.x;
  int ow = (int)(idx % 56); size_t t = idx / 56;
  int oh = (int)(t % 56);   t /= 56;
  size_t nc = t;                        // img*64 + c
  float best = -3.4e38f;
  for (int r = 0; r < 3; ++r) {
    int ih = oh * 2 - 1 + r;
    if ((unsigned)ih >= 112u) continue;
    for (int c2 = 0; c2 < 3; ++c2) {
      int iw = ow * 2 - 1 + c2;
      if ((unsigned)iw >= 112u) continue;
      best = fmaxf(best, bf2f(in[(nc * 112 + ih) * 112 + iw]));
    }
  }
  out[idx] = f2bf(best);
}

// global average pool over 7x7: [64,512,7,7] bf16 -> [64,512] bf16
__global__ void avgpool_k(const unsigned short* __restrict__ in, unsigned short* out) {
  int idx = blockIdx.x * 256 + threadIdx.x;        // 64*512
  const unsigned short* p = in + (size_t)idx * 49;
  float s = 0.f;
#pragma unroll
  for (int i = 0; i < 49; ++i) s += bf2f(p[i]);
  out[idx] = f2bf(s * (1.f / 49.f));
}

// FC2 (128->4) + bias + tanh * pi/2 -> angles [64,4] fp32
__global__ void fc2_k(const unsigned short* __restrict__ h, const float* __restrict__ w2,
                      const float* b2, float* angles) {
  int idx = threadIdx.x;                            // 256 = 64*4
  int co = idx & 3, img = idx >> 2;
  float s = b2[co];
  for (int k = 0; k < 128; ++k) s += bf2f(h[img * 128 + k]) * w2[co * 128 + k];
  angles[idx] = tanhf(s) * (PI_F * 0.5f);
}

// ---------------------------------------------------------------------------
// 4-qubit state-vector sim (wire w -> bit value 8>>w, q0 most significant)
// ---------------------------------------------------------------------------
__device__ __forceinline__ void q_ry(float* re, float* im, int v, float t) {
  float c = cosf(t * 0.5f), s = sinf(t * 0.5f);
#pragma unroll
  for (int b0 = 0; b0 < 16; ++b0) if (!(b0 & v)) {
    int b1 = b0 | v;
    float r0 = re[b0], i0 = im[b0], r1 = re[b1], i1 = im[b1];
    re[b0] = c * r0 - s * r1;  im[b0] = c * i0 - s * i1;
    re[b1] = s * r0 + c * r1;  im[b1] = s * i0 + c * i1;
  }
}
__device__ __forceinline__ void q_rz(float* re, float* im, int v, float t) {
  float c = cosf(t * 0.5f), s = sinf(t * 0.5f);
#pragma unroll
  for (int b0 = 0; b0 < 16; ++b0) if (!(b0 & v)) {
    int b1 = b0 | v;
    float r = re[b0], i = im[b0];
    re[b0] = r * c + i * s;  im[b0] = i * c - r * s;     // * e^{-i t/2}
    r = re[b1]; i = im[b1];
    re[b1] = r * c - i * s;  im[b1] = i * c + r * s;     // * e^{+i t/2}
  }
}
__device__ __forceinline__ void q_rx(float* re, float* im, int v, float t) {
  float c = cosf(t * 0.5f), s = sinf(t * 0.5f);
#pragma unroll
  for (int b0 = 0; b0 < 16; ++b0) if (!(b0 & v)) {
    int b1 = b0 | v;
    float r0 = re[b0], i0 = im[b0], r1 = re[b1], i1 = im[b1];
    re[b0] = c * r0 + s * i1;  im[b0] = c * i0 - s * r1;
    re[b1] = c * r1 + s * i0;  im[b1] = c * i1 - s * r0;
  }
}
__device__ __forceinline__ void q_cnot(float* re, float* im, int cv, int tv) {
#pragma unroll
  for (int b = 0; b < 16; ++b) if ((b & cv) && !(b & tv)) {
    int b2 = b | tv;
    float r = re[b]; re[b] = re[b2]; re[b2] = r;
    float i = im[b]; im[b] = im[b2]; im[b2] = i;
  }
}

__global__ void quantum_k(const float* __restrict__ angles, const float* __restrict__ qpin,
                          float* __restrict__ out) {
  int img = threadIdx.x;                 // 64 threads, one sample each
  if (img >= 64) return;
  float re[16], im[16];
#pragma unroll
  for (int i = 0; i < 16; ++i) { re[i] = 0.f; im[i] = 0.f; }
  re[0] = 1.f;
  float qp[36];
#pragma unroll
  for (int i = 0; i < 36; ++i)
    qp[i] = fminf(fmaxf(qpin[i], -2.f * PI_F), 2.f * PI_F);
  // encoding RY(angle_i) on wire i
#pragma unroll
  for (int w = 0; w < 4; ++w) {
    float a = fminf(fmaxf(angles[img * 4 + w], -PI_F), PI_F);
    q_ry(re, im, 8 >> w, a);
  }
  int idx = 0;
#pragma unroll
  for (int l = 0; l < 3; ++l) {
#pragma unroll
    for (int w = 0; w < 4; ++w) q_ry(re, im, 8 >> w, qp[idx++]);
    q_cnot(re, im, 8, 2);   // CNOT(0,2)
    q_cnot(re, im, 4, 1);   // CNOT(1,3)
#pragma unroll
    for (int w = 0; w < 4; ++w) q_rz(re, im, 8 >> w, qp[idx++]);
#pragma unroll
    for (int w = 0; w < 4; ++w) q_rx(re, im, 8 >> w, qp[idx++]);
  }
  float pb[16], s = 0.f;
#pragma unroll
  for (int b = 0; b < 16; ++b) { pb[b] = re[b]*re[b] + im[b]*im[b] + 1e-10f; s += pb[b]; }
  float inv = 1.f / s;
#pragma unroll
  for (int b = 0; b < 16; ++b) out[img * 16 + b] = pb[b] * inv;
}

// fp32 -> bf16 weight conversion
__global__ void cvt_k(const float* __restrict__ src, unsigned short* __restrict__ dst, int n) {
  int i = blockIdx.x * 256 + threadIdx.x;
  if (i < n) dst[i] = f2bf(src[i]);
}

// ---------------------------------------------------------------------------
// Host orchestration
// ---------------------------------------------------------------------------
extern "C" void kernel_launch(void* const* d_in, const int* in_sizes, int n_in,
                              void* d_out, int out_size, void* d_ws, size_t ws_size,
                              hipStream_t stream) {
  (void)in_sizes; (void)n_in; (void)out_size; (void)ws_size;
  const float* x  = (const float*)d_in[0];
  const float* qp = (const float*)d_in[1];

  // --- input index map (dict-insertion-order recursive flattening) ----------
  int ii = 2;
  const int conv1_w = ii++;
  const int bn1 = ii; ii += 4;
  struct BI { int w1, bn1, w2, bn2, dw, dbn; } blk[8];
  const int chans[5] = {64, 64, 128, 256, 512};
  int nb = 0;
  for (int si = 0; si < 4; ++si)
    for (int bi = 0; bi < 2; ++bi) {
      BI b;
      b.w1 = ii++; b.bn1 = ii; ii += 4;
      b.w2 = ii++; b.bn2 = ii; ii += 4;
      if (si > 0 && bi == 0) { b.dw = ii++; b.dbn = ii; ii += 4; }
      else                   { b.dw = -1;  b.dbn = -1; }
      blk[nb++] = b;
    }
  const int fr_w1 = ii++; const int fr_b1 = ii++;
  const int fr_bn = ii; ii += 4;
  const int fr_w2 = ii++; const int fr_b2 = ii++;

  // --- workspace bump allocator --------------------------------------------
  size_t off = 0;
  auto alloc = [&](size_t bytes) -> void* {
    void* p = (char*)d_ws + off;
    off += (bytes + 255) & ~(size_t)255;
    return p;
  };
  auto cvt = [&](int idx, size_t n) -> unsigned short* {
    unsigned short* dst = (unsigned short*)alloc(n * 2);
    cvt_k<<<(unsigned)((n + 255) / 256), 256, 0, stream>>>((const float*)d_in[idx], dst, (int)n);
    return dst;
  };

  // weights -> bf16
  unsigned short *wb1[8], *wb2[8], *wbd[8];
  nb = 0;
  for (int si = 0; si < 4; ++si) {
    int cout = chans[si + 1];
    for (int bi = 0; bi < 2; ++bi) {
      int cin = (bi == 0) ? chans[si] : cout;
      wb1[nb] = cvt(blk[nb].w1, (size_t)cout * cin * 9);
      wb2[nb] = cvt(blk[nb].w2, (size_t)cout * cout * 9);
      wbd[nb] = (blk[nb].dw >= 0) ? cvt(blk[nb].dw, (size_t)cout * cin) : nullptr;
      ++nb;
    }
  }
  unsigned short* w1b = cvt(fr_w1, (size_t)128 * 512);

  // activations (bf16)
  unsigned short* actP = (unsigned short*)alloc((size_t)64 * 64 * 112 * 112 * 2);
  unsigned short* pX   = (unsigned short*)alloc((size_t)64 * 64 * 56 * 56 * 2);
  unsigned short* pY   = (unsigned short*)alloc((size_t)64 * 64 * 56 * 56 * 2);
  unsigned short* pZ   = (unsigned short*)alloc((size_t)64 * 64 * 56 * 56 * 2);
  unsigned short* feats = (unsigned short*)alloc((size_t)64 * 512 * 2);
  unsigned short* hbuf  = (unsigned short*)alloc((size_t)64 * 128 * 2);
  float* angles         = (float*)alloc(256 * 4);

  auto launch_conv = [&](const unsigned short* in, const unsigned short* w, int bnb,
                         const float* bias, const unsigned short* res, unsigned short* out,
                         int Cin, int H, int Cout, int OH, int stride, int KH, int relu) {
    ConvP p;
    p.in = in; p.w = w;
    p.bng = (const float*)d_in[bnb];     p.bnb = (const float*)d_in[bnb + 1];
    p.bnm = (const float*)d_in[bnb + 2]; p.bnv = (const float*)d_in[bnb + 3];
    p.bias = bias; p.res = res; p.out = out;
    p.Cin = Cin; p.H = H; p.Cout = Cout; p.OH = OH; p.relu = relu;
    int M = 64 * OH * OH;
    if (Cout >= 128) {
      dim3 grid((unsigned)(M / TM), (unsigned)(Cout / 128));
      const size_t lds = 4096 + 128 * 64;          // 12 KB
      if (KH == 3 && stride == 1)
        conv_wmma_k<3, 1, 1, 128><<<grid, 256, lds, stream>>>(p);
      else if (KH == 3)
        conv_wmma_k<3, 2, 1, 128><<<grid, 256, lds, stream>>>(p);
      else if (stride == 1)
        conv_wmma_k<1, 1, 0, 128><<<grid, 256, lds, stream>>>(p);
      else
        conv_wmma_k<1, 2, 0, 128><<<grid, 256, lds, stream>>>(p);
    } else {
      dim3 grid((unsigned)(M / TM), (unsigned)(Cout / 64));
      const size_t lds = 4096 + 64 * 64;           // 8 KB
      if (KH == 3)
        conv_wmma_k<3, 1, 1, 64><<<grid, 256, lds, stream>>>(p);
      else
        conv_wmma_k<1, 1, 0, 64><<<grid, 256, lds, stream>>>(p);
    }
  };

  // --- stem -----------------------------------------------------------------
  conv1_k<<<(unsigned)(((size_t)64 * 64 * 112 * 112) / 256), 256, 0, stream>>>(
      x, (const float*)d_in[conv1_w],
      (const float*)d_in[bn1], (const float*)d_in[bn1 + 1],
      (const float*)d_in[bn1 + 2], (const float*)d_in[bn1 + 3], actP);
  maxpool_k<<<(unsigned)(((size_t)64 * 64 * 56 * 56) / 256), 256, 0, stream>>>(actP, pX);

  // --- residual stages ------------------------------------------------------
  int curC = 64, curH = 56;
  nb = 0;
  for (int si = 0; si < 4; ++si) {
    int cout = chans[si + 1];
    for (int bi = 0; bi < 2; ++bi, ++nb) {
      int stride = (si > 0 && bi == 0) ? 2 : 1;
      int oh = curH / stride;
      // conv1 + bn1 + relu : X -> Y
      launch_conv(pX, wb1[nb], blk[nb].bn1, nullptr, nullptr, pY,
                  curC, curH, cout, oh, stride, 3, 1);
      if (blk[nb].dw >= 0) {
        // downsample: X -> Z (bn, no relu)
        launch_conv(pX, wbd[nb], blk[nb].dbn, nullptr, nullptr, pZ,
                    curC, curH, cout, oh, stride, 1, 0);
        // conv2 + bn2 + residual(Z) + relu : Y -> X
        launch_conv(pY, wb2[nb], blk[nb].bn2, nullptr, pZ, pX,
                    cout, oh, cout, oh, 1, 3, 1);
      } else {
        // conv2 + bn2 + residual(X) + relu : Y -> Z, then X := Z
        launch_conv(pY, wb2[nb], blk[nb].bn2, nullptr, pX, pZ,
                    cout, oh, cout, oh, 1, 3, 1);
        unsigned short* t = pX; pX = pZ; pZ = t;
      }
      curC = cout; curH = oh;
    }
  }

  // --- head -----------------------------------------------------------------
  avgpool_k<<<128, 256, 0, stream>>>(pX, feats);                       // [64,512]
  // FC1 (512->128) + b1 + BN + ReLU via the WMMA GEMM path (1x1, H=W=1)
  launch_conv(feats, w1b, fr_bn, (const float*)d_in[fr_b1], nullptr, hbuf,
              512, 1, 128, 1, 1, 1, 1);
  fc2_k<<<1, 256, 0, stream>>>(hbuf, (const float*)d_in[fr_w2],
                               (const float*)d_in[fr_b2], angles);
  quantum_k<<<1, 64, 0, stream>>>(angles, qp, (float*)d_out);
}
